// GNN_73383811219611
// MI455X (gfx1250) — compile-verified, matrix operand order checked by reading, not run on hardware
//
#include <hip/hip_runtime.h>

// GraphSAGE 3-layer + BN + final linear for MI455X (gfx1250, wave32).
// All GEMM operands staged in f16 memory -> inner loop is pure
// global_load_b128 + v_wmma_f32_16x16x32_f16 (f32 accumulate, zero cvt VALU).
// Edge scatter via f32 L2 atomics (node features are L2-resident).

#define C128 128
#define BN_EPS 1e-5f

typedef __attribute__((ext_vector_type(16))) _Float16 v16h;
typedef __attribute__((ext_vector_type(8)))  _Float16 v8h;
typedef __attribute__((ext_vector_type(4)))  _Float16 v4h;
typedef __attribute__((ext_vector_type(8)))  float    v8f;

// ---------------------------------------------------------------------------
// Load one lane's 32-wide K slice of a 16-bit WMMA operand (CDNA5 ISA 7.12.2):
//   lane l: matrix row = (l & 15); kb = (l >= 16) ? 8 : 0
//   halves 0..7  <- k = kb    .. kb+7     (16B contiguous -> b128)
//   halves 8..15 <- k = 16+kb .. 16+kb+7  (16B contiguous -> b128)
// ---------------------------------------------------------------------------
__device__ __forceinline__ v16h load_k32_h(const _Float16* p, int kb) {
  const v8h lo = *(const v8h*)(p + kb);
  const v8h hi = *(const v8h*)(p + kb + 16);
  return __builtin_shufflevector(lo, hi, 0, 1, 2, 3, 4, 5, 6, 7, 8, 9, 10, 11,
                                 12, 13, 14, 15);
}

// ---------------------------------------------------------------------------
// Fused GEMM: Y = relu( sum_{p<NP} Ap @ Bp^T + bias ), each part K=128, f16.
//   Bp is (128, K=128) f16 row-major with row stride ldb (the ISA B layout for
//   D=A*B equals the A layout of B^T = W row-major -> no transpose needed).
//   STATS: atomically accumulate per-column sum / sum^2 of relu'd output into
//   stats[0..127] / stats[128..255] (BatchNorm pass 1).
// Block: 256 threads = 8 waves; 64 rows x 128 cols; wave w owns columns
// [16w,16w+16) across 4 row tiles. Out-of-range A rows are clamped (their
// results are never stored), so the K loop has zero predication.
// ---------------------------------------------------------------------------
template <int NP, bool STATS>
__global__ __launch_bounds__(256) void gemm_wmma_fused(
    const _Float16* __restrict__ A0, const _Float16* __restrict__ A1,
    const _Float16* __restrict__ A2,
    const _Float16* __restrict__ B0, const _Float16* __restrict__ B1,
    const _Float16* __restrict__ B2, int ldb,
    const float* __restrict__ bias,
    float* __restrict__ Y, float* stats, int nRows) {
  const int lane    = threadIdx.x & 31;
  const int wave    = threadIdx.x >> 5;
  const int rowBase = blockIdx.x * 64;
  const int lrow    = lane & 15;
  const int kb      = (lane & 16) ? 8 : 0;
  const int bcol    = wave * 16 + lrow;  // output column / B row

  const _Float16* Ap[3] = {A0, A1, A2};
  const _Float16* Bp[3] = {B0, B1, B2};

  // Clamped A-row offsets (in halves); no load predication anywhere.
  size_t aoff[4];
#pragma unroll
  for (int rt = 0; rt < 4; ++rt) {
    int r = rowBase + rt * 16 + lrow;
    r = (r < nRows) ? r : (nRows - 1);
    aoff[rt] = (size_t)r * C128;
  }

  v8f acc[4];
#pragma unroll
  for (int i = 0; i < 4; ++i)
#pragma unroll
    for (int j = 0; j < 8; ++j) acc[i][j] = 0.f;

#pragma unroll
  for (int p = 0; p < NP; ++p) {
    const _Float16* a    = Ap[p];
    const _Float16* brow = Bp[p] + (size_t)bcol * ldb;
#pragma unroll
    for (int kc = 0; kc < 128; kc += 32) {
      const v16h breg = load_k32_h(brow + kc, kb);
#pragma unroll
      for (int rt = 0; rt < 4; ++rt) {
        const v16h areg = load_k32_h(a + aoff[rt] + kc, kb);
        acc[rt] = __builtin_amdgcn_wmma_f32_16x16x32_f16(
            false, areg, false, breg, (short)0, acc[rt], false, false);
      }
    }
  }

  // Epilogue. D layout: lane l -> N = (l&15); VGPR v -> M = v + 8*(l>>4).
  const float bval  = bias[bcol];
  const int   mbase = (lane >> 4) << 3;
  float psum = 0.f, psq = 0.f;

  if (rowBase + 64 <= nRows) {  // fast path: every block except the last
#pragma unroll
    for (int rt = 0; rt < 4; ++rt) {
      float* yrow = Y + (size_t)(rowBase + rt * 16 + mbase) * C128 + bcol;
#pragma unroll
      for (int v = 0; v < 8; ++v) {
        const float val = fmaxf(acc[rt][v] + bval, 0.f);
        yrow[(size_t)v * C128] = val;
        psum += val;
        psq  += val * val;
      }
    }
  } else {
#pragma unroll
    for (int rt = 0; rt < 4; ++rt) {
      const int rb = rowBase + rt * 16 + mbase;
#pragma unroll
      for (int v = 0; v < 8; ++v) {
        const int row = rb + v;
        if (row < nRows) {
          const float val = fmaxf(acc[rt][v] + bval, 0.f);
          Y[(size_t)row * C128 + bcol] = val;
          psum += val;
          psq  += val * val;
        }
      }
    }
  }

  if (STATS) {
    psum += __shfl_xor(psum, 16, 32);  // pair half-column lanes (M<8 | M>=8)
    psq  += __shfl_xor(psq, 16, 32);
    if (lane < 16) {
      atomicAdd(&stats[bcol], psum);
      atomicAdd(&stats[C128 + bcol], psq);
    }
  }
}

// ---------------------------------------------------------------------------
// Edge kernels
// ---------------------------------------------------------------------------
__global__ void deg_count_kernel(const long long* __restrict__ dst,
                                 float* deg, int E) {
  const int e = blockIdx.x * blockDim.x + threadIdx.x;
  if (e < E) atomicAdd(&deg[(int)dst[e]], 1.0f);
}

__global__ void deg_inv_kernel(float* deg, int n) {
  const int i = blockIdx.x * blockDim.x + threadIdx.x;
  if (i < n) deg[i] = 1.0f / fmaxf(deg[i], 1.0f);
}

// One wave per edge: lane g moves channels [4g, 4g+4): 8B f16 gather ->
// 4 f32 L2 atomic adds. Node features are L2-resident (12.8 MB << 192 MB).
__global__ void scatter_add_kernel(const _Float16* __restrict__ h,
                                   const long long* __restrict__ src,
                                   const long long* __restrict__ dst,
                                   float* agg, int E) {
  const long long id = (long long)blockIdx.x * blockDim.x + threadIdx.x;
  if (id >= (long long)E * 32) return;
  const int e = (int)(id >> 5);
  const int g = (int)(id & 31);
  const int s = (int)src[e];
  const int d = (int)dst[e];
  const v4h v = *(const v4h*)(h + (size_t)s * C128 + g * 4);
  float* o = agg + (size_t)d * C128 + g * 4;
  atomicAdd(o + 0, (float)v[0]);
  atomicAdd(o + 1, (float)v[1]);
  atomicAdd(o + 2, (float)v[2]);
  atomicAdd(o + 3, (float)v[3]);
}

// ---------------------------------------------------------------------------
// Elementwise helpers
// ---------------------------------------------------------------------------
__global__ void f32_to_f16_kernel(const float* __restrict__ s,
                                  _Float16* __restrict__ d, long long n) {
  const long long i = (long long)blockIdx.x * blockDim.x + threadIdx.x;
  if (i < n) d[i] = (_Float16)s[i];
}

// agg16[n][c] = f16( agg[n][c] * scale[n] )  (mean-aggregation divide, folded
// out of the GEMM hot loop: (S*A)@W == S*(A@W) row-wise)
__global__ void aggscale_cvt_kernel(const float* __restrict__ agg,
                                    const float* __restrict__ scale,
                                    _Float16* __restrict__ agg16,
                                    long long total) {
  const long long i = (long long)blockIdx.x * blockDim.x + threadIdx.x;
  if (i < total) agg16[i] = (_Float16)(agg[i] * scale[i >> 7]);
}

__global__ void bn_finalize_kernel(const float* __restrict__ stats,
                                   const float* __restrict__ g,
                                   const float* __restrict__ b,
                                   float* bnp, float invN) {
  const int c = threadIdx.x;  // 128 threads
  const float mu  = stats[c] * invN;
  const float var = fmaxf(stats[C128 + c] * invN - mu * mu, 0.f);
  const float sc  = g[c] * rsqrtf(var + BN_EPS);
  bnp[c]        = sc;
  bnp[C128 + c] = b[c] - mu * sc;
}

// h16 = f16( y*sc + sh )
__global__ void bn_apply_kernel(const float* __restrict__ y,
                                const float* __restrict__ bnp,
                                _Float16* __restrict__ h, long long total) {
  const long long i = (long long)blockIdx.x * blockDim.x + threadIdx.x;
  if (i < total) {
    const int c = (int)(i & (C128 - 1));
    h[i] = (_Float16)(y[i] * bnp[c] + bnp[C128 + c]);
  }
}

__global__ void zero_kernel(float* p, long long n) {
  const long long i = (long long)blockIdx.x * blockDim.x + threadIdx.x;
  if (i < n) p[i] = 0.f;
}

// ---------------------------------------------------------------------------
extern "C" void kernel_launch(void* const* d_in, const int* in_sizes, int n_in,
                              void* d_out, int out_size, void* d_ws,
                              size_t ws_size, hipStream_t stream) {
  const float*     x  = (const float*)d_in[0];
  const long long* ei = (const long long*)d_in[1];  // int64 (2, E)
  const int N = in_sizes[0] / C128;
  const int E = in_sizes[1] / 2;
  const long long* src = ei;
  const long long* dst = ei + E;

  const float* Wl[3] = {(const float*)d_in[2], (const float*)d_in[7],
                        (const float*)d_in[12]};
  const float* bl[3] = {(const float*)d_in[3], (const float*)d_in[8],
                        (const float*)d_in[13]};
  const float* Wr[3] = {(const float*)d_in[4], (const float*)d_in[9],
                        (const float*)d_in[14]};
  const float* gg[3] = {(const float*)d_in[5], (const float*)d_in[10],
                        (const float*)d_in[15]};
  const float* bb[3] = {(const float*)d_in[6], (const float*)d_in[11],
                        (const float*)d_in[16]};
  const float* Wlin = (const float*)d_in[17];
  const float* blin = (const float*)d_in[18];

  // Workspace (~90 MB):
  //   f32: agg/y | scale | stats | bnp
  //   f16: x16 | agg16 | h16[3] | Wl16[3],Wr16[3] | Wlin16
  float* ws = (float*)d_ws;
  const size_t NC = (size_t)N * C128;
  float* agg   = ws;
  float* scale = agg + NC;
  float* stats = scale + N;
  float* bnp   = stats + 256;
  _Float16* x16   = (_Float16*)(bnp + 256);
  _Float16* agg16 = x16 + NC;
  _Float16* h16[3] = {agg16 + NC, agg16 + 2 * NC, agg16 + 3 * NC};
  _Float16* wl16[3], *wr16[3];
  _Float16* wp = agg16 + 4 * NC;
  for (int l = 0; l < 3; ++l) {
    wl16[l] = wp; wp += C128 * C128;
    wr16[l] = wp; wp += C128 * C128;
  }
  _Float16* wlin16 = wp;  // 128 x 384

  const int TPB = 256;
  const unsigned gN    = (unsigned)((N + TPB - 1) / TPB);
  const unsigned gE    = (unsigned)((E + TPB - 1) / TPB);
  const unsigned gNC   = (unsigned)((NC + TPB - 1) / TPB);
  const unsigned gW    = (unsigned)((C128 * C128 + TPB - 1) / TPB);
  const unsigned gWlin = (unsigned)((C128 * 384 + TPB - 1) / TPB);
  const unsigned gScat = (unsigned)(((long long)E * 32 + TPB - 1) / TPB);
  const unsigned gGemm = (unsigned)((N + 63) / 64);

  // One-time prep: degree -> 1/max(deg,1); weights and x to f16.
  zero_kernel<<<gN, TPB, 0, stream>>>(scale, (long long)N);
  deg_count_kernel<<<gE, TPB, 0, stream>>>(dst, scale, E);
  deg_inv_kernel<<<gN, TPB, 0, stream>>>(scale, N);
  for (int l = 0; l < 3; ++l) {
    f32_to_f16_kernel<<<gW, TPB, 0, stream>>>(Wl[l], wl16[l], C128 * C128);
    f32_to_f16_kernel<<<gW, TPB, 0, stream>>>(Wr[l], wr16[l], C128 * C128);
  }
  f32_to_f16_kernel<<<gWlin, TPB, 0, stream>>>(Wlin, wlin16, C128 * 384);
  f32_to_f16_kernel<<<gNC, TPB, 0, stream>>>(x, x16, (long long)NC);

  const _Float16* hin = x16;
  for (int l = 0; l < 3; ++l) {
    zero_kernel<<<gNC, TPB, 0, stream>>>(agg, (long long)NC);
    scatter_add_kernel<<<gScat, TPB, 0, stream>>>(hin, src, dst, agg, E);
    aggscale_cvt_kernel<<<gNC, TPB, 0, stream>>>(agg, scale, agg16,
                                                 (long long)NC);
    zero_kernel<<<1, 256, 0, stream>>>(stats, 256);
    // y = relu([agg16 | hin] @ [Wl|Wr]^T + bl) -> agg (f32), + BN stats
    gemm_wmma_fused<2, true><<<gGemm, TPB, 0, stream>>>(
        agg16, hin, nullptr, wl16[l], wr16[l], nullptr, C128, bl[l], agg,
        stats, N);
    bn_finalize_kernel<<<1, 128, 0, stream>>>(stats, gg[l], bb[l], bnp,
                                              1.0f / (float)N);
    bn_apply_kernel<<<gNC, TPB, 0, stream>>>(agg, bnp, h16[l], (long long)NC);
    hin = h16[l];
  }

  // out = relu([h1|h2|h3] @ Wlin^T + blin), Wlin16 is (128, 384) row-major
  gemm_wmma_fused<3, false><<<gGemm, TPB, 0, stream>>>(
      h16[0], h16[1], h16[2], wlin16, wlin16 + 128, wlin16 + 256, 384, blin,
      (float*)d_out, nullptr, N);
}
